// CHARM_10677288698628
// MI455X (gfx1250) — compile-verified, matrix operand order checked by reading, not run on hardware
//
#include <hip/hip_runtime.h>

// ---------------------------------------------------------------------------
// GNN message-passing network on gfx1250 via v_wmma_f32_16x16x32_bf16.
// N=20000 nodes, E=320000 edges, NODE_IN=64, EDGE_DIM=32, HID=128, L=3.
// Weights staged into LDS with GLOBAL_LOAD_ASYNC_TO_LDS_B128 (ASYNCcnt).
// ---------------------------------------------------------------------------

typedef __bf16 bf16_t;
typedef __attribute__((ext_vector_type(16))) __bf16 v16bf;
typedef __attribute__((ext_vector_type(8)))  __bf16 v8bf;
typedef __attribute__((ext_vector_type(8)))  float  v8f;

#define HID  128
#define EDIM 32
#define NIN  64
#define NLAYER 3
#define WPB 8          // waves per 256-thread block (wave32)

// ---- helpers --------------------------------------------------------------

// native bf16 convert (v_cvt_pk_bf16_f32 on gfx1250, RNE)
static __device__ __forceinline__ bf16_t f2bf(float f) { return (bf16_t)f; }

static __device__ __forceinline__ v16bf combine16(v8bf lo, v8bf hi) {
    union { v16bf v; v8bf h[2]; } u;
    u.h[0] = lo; u.h[1] = hi;
    return u.v;
}

static __device__ __forceinline__ v8f vzero8() {
    v8f z;
#pragma unroll
    for (int i = 0; i < 8; ++i) z[i] = 0.0f;
    return z;
}

// A-fragment (16x32 bf16): lane holds row M=lane&15; two contiguous 8-elem
// K-chunks at koff=8*half and 16+8*half.  rowk = row base + kstep*32.
static __device__ __forceinline__ v16bf a_frag_bf16(const bf16_t* rowk, int half) {
    const v8bf* p0 = (const v8bf*)(rowk + half * 8);
    const v8bf* p1 = (const v8bf*)(rowk + 16 + half * 8);
    return combine16(*p0, *p1);
}

static __device__ __forceinline__ v8bf cvt8(const float* p) {
    const float4* q = (const float4*)p;
    float4 a = q[0], b = q[1];
    v8bf r;
    r[0] = f2bf(a.x); r[1] = f2bf(a.y); r[2] = f2bf(a.z); r[3] = f2bf(a.w);
    r[4] = f2bf(b.x); r[5] = f2bf(b.y); r[6] = f2bf(b.z); r[7] = f2bf(b.w);
    return r;
}

static __device__ __forceinline__ v16bf a_frag_f32(const float* rowk, int half) {
    return combine16(cvt8(rowk + half * 8), cvt8(rowk + 16 + half * 8));
}

// B-fragment from pre-swizzled storage: ((frag*32 + lane)*16) contiguous bf16.
static __device__ __forceinline__ v16bf b_frag(const bf16_t* frags, int fragIdx, int lane) {
    const bf16_t* p = frags + ((size_t)fragIdx * 32 + lane) * 16;
    return combine16(*(const v8bf*)p, *(const v8bf*)(p + 8));
}

static __device__ __forceinline__ v8f wmma_bf16(v16bf a, v16bf b, v8f c) {
    return __builtin_amdgcn_wmma_f32_16x16x32_bf16(
        /*neg_a=*/false, a, /*neg_b=*/false, b,
        /*c_mod=*/(short)0, c, /*reuse_a=*/false, /*reuse_b=*/false);
}

static __device__ __forceinline__ void copy_to_lds(unsigned* dst, const unsigned* src, int nwords) {
    for (int i = threadIdx.x; i < nwords; i += blockDim.x) dst[i] = src[i];
}

// LDS byte address = low 32 bits of the generic (shared-aperture) pointer.
static __device__ __forceinline__ unsigned lds_addr_u32(const void* p) {
    return (unsigned)(unsigned long long)p;
}

// Async DMA global -> LDS, 16B per thread per issue; tracked by ASYNCcnt.
// GVS mode: mem = SADDR(64) + VADDR(32); LDS addr from VDST VGPR.
static __device__ __forceinline__ void async_copy_to_lds(void* ldsDst, const void* gSrc,
                                                         int nbytes) {
    unsigned ldsBase = lds_addr_u32(ldsDst);
    int nchunks = nbytes >> 4;
    for (int i = threadIdx.x; i < nchunks; i += blockDim.x) {
        unsigned lo = ldsBase + (unsigned)(i << 4);
        unsigned go = (unsigned)(i << 4);
        asm volatile("global_load_async_to_lds_b128 %0, %1, %2"
                     :: "v"(lo), "v"(go), "s"(gSrc) : "memory");
    }
    asm volatile("s_wait_asynccnt 0x0" ::: "memory");
}

// ---- weight swizzle: fp32 [K][N] -> bf16 WMMA B-fragment layout -----------
// out[((ks*NT + nt)*32 + lane)*16 + j] = W[ks*32 + kidx(j,lane)][nt*16 + lane%16]
__global__ void k_swizzle(const float* __restrict__ W, bf16_t* __restrict__ out,
                          int K, int Ncols) {
    int total = K * Ncols;
    int NT = Ncols >> 4;
    for (int idx = blockIdx.x * blockDim.x + threadIdx.x; idx < total;
         idx += gridDim.x * blockDim.x) {
        int frag = idx >> 9;          // 32 lanes * 16 elems
        int rem  = idx & 511;
        int lane = rem >> 4;
        int j    = rem & 15;
        int nt   = frag % NT;
        int ks   = frag / NT;
        int kid  = ks * 32 + j + 8 * ((j >> 3) + (lane >> 4));
        int col  = nt * 16 + (lane & 15);
        out[idx] = f2bf(W[kid * Ncols + col]);
    }
}

// ---- node projection: h = x @ node_w + node_b  ([N,64]x[64,128]) ----------
__global__ __launch_bounds__(256)
void k_node_proj(const float* __restrict__ x, const bf16_t* __restrict__ wS,
                 const float* __restrict__ bias, bf16_t* __restrict__ h, int ntiles) {
    __shared__ __align__(16) bf16_t sw[NIN * HID];
    copy_to_lds((unsigned*)sw, (const unsigned*)wS, NIN * HID / 2);
    __syncthreads();

    int wave = threadIdx.x >> 5, lane = threadIdx.x & 31;
    int half = lane >> 4, r = lane & 15;

    for (int tile = blockIdx.x * WPB + wave; tile < ntiles; tile += gridDim.x * WPB) {
        const float* row = x + (size_t)(tile * 16 + r) * NIN;
        v8f acc[8];
#pragma unroll
        for (int nt = 0; nt < 8; ++nt) acc[nt] = vzero8();
#pragma unroll
        for (int ks = 0; ks < 2; ++ks) {
            v16bf a = a_frag_f32(row + ks * 32, half);
#pragma unroll
            for (int nt = 0; nt < 8; ++nt)
                acc[nt] = wmma_bf16(a, b_frag(sw, ks * 8 + nt, lane), acc[nt]);
        }
#pragma unroll
        for (int nt = 0; nt < 8; ++nt) {
            int col = nt * 16 + r;
            float bv = bias[col];
#pragma unroll
            for (int v = 0; v < 8; ++v) {
                int orow = tile * 16 + v + 8 * half;
                h[(size_t)orow * HID + col] = f2bf(acc[nt][v] + bv);
            }
        }
    }
}

// ---- edge projection: e = edge_attr @ edge_w + edge_b  ([E,32]x[32,32]) ---
__global__ __launch_bounds__(256)
void k_edge_proj(const float* __restrict__ ea, const bf16_t* __restrict__ wS,
                 const float* __restrict__ bias, bf16_t* __restrict__ e, int ntiles) {
    __shared__ __align__(16) bf16_t sw[EDIM * EDIM];
    copy_to_lds((unsigned*)sw, (const unsigned*)wS, EDIM * EDIM / 2);
    __syncthreads();

    int wave = threadIdx.x >> 5, lane = threadIdx.x & 31;
    int half = lane >> 4, r = lane & 15;

    for (int tile = blockIdx.x * WPB + wave; tile < ntiles; tile += gridDim.x * WPB) {
        const float* row = ea + (size_t)(tile * 16 + r) * EDIM;
        v8f acc[2];
        acc[0] = vzero8(); acc[1] = vzero8();
        v16bf a = a_frag_f32(row, half);
#pragma unroll
        for (int nt = 0; nt < 2; ++nt)
            acc[nt] = wmma_bf16(a, b_frag(sw, nt, lane), acc[nt]);
#pragma unroll
        for (int nt = 0; nt < 2; ++nt) {
            int col = nt * 16 + r;
            float bv = bias[col];
#pragma unroll
            for (int v = 0; v < 8; ++v) {
                int orow = tile * 16 + v + 8 * half;
                e[(size_t)orow * EDIM + col] = f2bf(acc[nt][v] + bv);
            }
        }
    }
}

// ---- edge message + scatter-add ------------------------------------------
// m = relu(relu([h[dst],h[src],e] @ W1 + b1) @ W2 + b2); agg[dst] += m
// Dynamic LDS: W1 frags (73728 B) | W2 frags (32768 B) | per-wave m1 (8*4096 B)
__global__ __launch_bounds__(256)
void k_msg(const bf16_t* __restrict__ h, const bf16_t* __restrict__ e,
           const int* __restrict__ srcIdx, const int* __restrict__ dstIdx,
           const bf16_t* __restrict__ w1S, const float* __restrict__ b1,
           const bf16_t* __restrict__ w2S, const float* __restrict__ b2,
           float* __restrict__ agg, int ntiles) {
    extern __shared__ __align__(16) char smem[];
    bf16_t* s_w1 = (bf16_t*)smem;                       // 9*8 frags
    bf16_t* s_w2 = (bf16_t*)(smem + 73728);             // 4*8 frags
    int wave = threadIdx.x >> 5, lane = threadIdx.x & 31;
    bf16_t* s_m1 = (bf16_t*)(smem + 106496) + wave * (16 * HID);

    async_copy_to_lds(s_w1, w1S, 73728);
    async_copy_to_lds(s_w2, w2S, 32768);
    __syncthreads();

    int half = lane >> 4, r = lane & 15;

    for (int tile = blockIdx.x * WPB + wave; tile < ntiles; tile += gridDim.x * WPB) {
        int eg = tile * 16 + r;
        int dn = dstIdx[eg];
        int sn = srcIdx[eg];
        const bf16_t* hd = h + (size_t)dn * HID;
        const bf16_t* hs = h + (size_t)sn * HID;
        const bf16_t* er = e + (size_t)eg * EDIM;

        v8f acc[8];
#pragma unroll
        for (int nt = 0; nt < 8; ++nt) acc[nt] = vzero8();

        // K = 288 : [h_dst(128) | h_src(128) | e(32)]
#pragma unroll
        for (int ks = 0; ks < 9; ++ks) {
            v16bf a;
            if (ks < 4)      a = a_frag_bf16(hd + ks * 32, half);
            else if (ks < 8) a = a_frag_bf16(hs + (ks - 4) * 32, half);
            else             a = a_frag_bf16(er, half);
#pragma unroll
            for (int nt = 0; nt < 8; ++nt)
                acc[nt] = wmma_bf16(a, b_frag(s_w1, ks * 8 + nt, lane), acc[nt]);
        }

        // relu(acc + b1) -> per-wave LDS tile (row-major 16x128 bf16)
#pragma unroll
        for (int nt = 0; nt < 8; ++nt) {
            int col = nt * 16 + r;
            float bv = b1[col];
#pragma unroll
            for (int v = 0; v < 8; ++v) {
                int mrow = v + 8 * half;
                s_m1[mrow * HID + col] = f2bf(fmaxf(acc[nt][v] + bv, 0.0f));
            }
        }
        asm volatile("s_wait_dscnt 0x0" ::: "memory");

        // second GEMM: m1[16x128] @ W2[128x128]
        v8f acc2[8];
#pragma unroll
        for (int nt = 0; nt < 8; ++nt) acc2[nt] = vzero8();
#pragma unroll
        for (int ks = 0; ks < 4; ++ks) {
            v16bf a = a_frag_bf16(s_m1 + r * HID + ks * 32, half);
#pragma unroll
            for (int nt = 0; nt < 8; ++nt)
                acc2[nt] = wmma_bf16(a, b_frag(s_w2, ks * 8 + nt, lane), acc2[nt]);
        }

        // relu + scatter-add into agg[dst]
#pragma unroll
        for (int v = 0; v < 8; ++v) {
            int drow = dstIdx[tile * 16 + v + 8 * half];
            float* abase = agg + (size_t)drow * HID;
#pragma unroll
            for (int nt = 0; nt < 8; ++nt) {
                int col = nt * 16 + r;
                float mv = fmaxf(acc2[nt][v] + b2[col], 0.0f);
                atomicAdd(abase + col, mv);
            }
        }
    }
}

// ---- node update: h = relu(relu([h,agg] @ U1 + c1) @ U2 + c2) -------------
// Dynamic LDS: U1 frags (65536 B) | U2 frags (32768 B) | per-wave m1 (8*4096 B)
__global__ __launch_bounds__(256)
void k_update(bf16_t* __restrict__ hio, const float* __restrict__ agg,
              const bf16_t* __restrict__ w1S, const float* __restrict__ b1,
              const bf16_t* __restrict__ w2S, const float* __restrict__ b2,
              float* __restrict__ hf32, int ntiles) {
    extern __shared__ __align__(16) char smem[];
    bf16_t* s_w1 = (bf16_t*)smem;                       // 8*8 frags
    bf16_t* s_w2 = (bf16_t*)(smem + 65536);             // 4*8 frags
    int wave = threadIdx.x >> 5, lane = threadIdx.x & 31;
    bf16_t* s_m1 = (bf16_t*)(smem + 98304) + wave * (16 * HID);

    async_copy_to_lds(s_w1, w1S, 65536);
    async_copy_to_lds(s_w2, w2S, 32768);
    __syncthreads();

    int half = lane >> 4, r = lane & 15;

    for (int tile = blockIdx.x * WPB + wave; tile < ntiles; tile += gridDim.x * WPB) {
        int node = tile * 16 + r;
        const bf16_t* hr = hio + (size_t)node * HID;
        const float*  ar = agg + (size_t)node * HID;

        v8f acc[8];
#pragma unroll
        for (int nt = 0; nt < 8; ++nt) acc[nt] = vzero8();

        // K = 256 : [h(128) | agg(128)]
#pragma unroll
        for (int ks = 0; ks < 8; ++ks) {
            v16bf a = (ks < 4) ? a_frag_bf16(hr + ks * 32, half)
                               : a_frag_f32(ar + (ks - 4) * 32, half);
#pragma unroll
            for (int nt = 0; nt < 8; ++nt)
                acc[nt] = wmma_bf16(a, b_frag(s_w1, ks * 8 + nt, lane), acc[nt]);
        }

#pragma unroll
        for (int nt = 0; nt < 8; ++nt) {
            int col = nt * 16 + r;
            float bv = b1[col];
#pragma unroll
            for (int v = 0; v < 8; ++v) {
                int mrow = v + 8 * half;
                s_m1[mrow * HID + col] = f2bf(fmaxf(acc[nt][v] + bv, 0.0f));
            }
        }
        asm volatile("s_wait_dscnt 0x0" ::: "memory");

        v8f acc2[8];
#pragma unroll
        for (int nt = 0; nt < 8; ++nt) acc2[nt] = vzero8();
#pragma unroll
        for (int ks = 0; ks < 4; ++ks) {
            v16bf a = a_frag_bf16(s_m1 + r * HID + ks * 32, half);
#pragma unroll
            for (int nt = 0; nt < 8; ++nt)
                acc2[nt] = wmma_bf16(a, b_frag(s_w2, ks * 8 + nt, lane), acc2[nt]);
        }

#pragma unroll
        for (int nt = 0; nt < 8; ++nt) {
            int col = nt * 16 + r;
            float bv = b2[col];
#pragma unroll
            for (int v = 0; v < 8; ++v) {
                int orow = tile * 16 + v + 8 * half;
                float hv = fmaxf(acc2[nt][v] + bv, 0.0f);
                hio[(size_t)orow * HID + col] = f2bf(hv);
                if (hf32) hf32[(size_t)orow * HID + col] = hv;
            }
        }
    }
}

// ---- token head (exact fp32): logits = relu(h@W1+b1)@W2+b2 ----------------
__global__ __launch_bounds__(256)
void k_token(const float* __restrict__ hf, const float* __restrict__ w1,
             const float* __restrict__ b1, const float* __restrict__ w2,
             const float* __restrict__ b2, float* __restrict__ logits, int n) {
    int node = blockIdx.x * blockDim.x + threadIdx.x;
    if (node >= n) return;
    const float* hr = hf + (size_t)node * HID;
    float t[64];
#pragma unroll
    for (int j = 0; j < 64; ++j) t[j] = b1[j];
    for (int k = 0; k < HID; ++k) {
        float hk = hr[k];
        const float* wr = w1 + k * 64;
#pragma unroll
        for (int j = 0; j < 64; ++j) t[j] = fmaf(hk, wr[j], t[j]);
    }
    float logit = b2[0];
#pragma unroll
    for (int j = 0; j < 64; ++j) logit = fmaf(fmaxf(t[j], 0.0f), w2[j], logit);
    logits[node] = logit;
}

// ---------------------------------------------------------------------------
extern "C" void kernel_launch(void* const* d_in, const int* in_sizes, int n_in,
                              void* d_out, int out_size, void* d_ws, size_t ws_size,
                              hipStream_t stream) {
    const float* x       = (const float*)d_in[0];
    const float* eattr   = (const float*)d_in[1];
    const float* node_w  = (const float*)d_in[2];
    const float* node_b  = (const float*)d_in[3];
    const float* edge_w  = (const float*)d_in[4];
    const float* edge_b  = (const float*)d_in[5];
    const float* msg_w1  = (const float*)d_in[6];
    const float* msg_b1  = (const float*)d_in[7];
    const float* msg_w2  = (const float*)d_in[8];
    const float* msg_b2  = (const float*)d_in[9];
    const float* up_w1   = (const float*)d_in[10];
    const float* up_b1   = (const float*)d_in[11];
    const float* up_w2   = (const float*)d_in[12];
    const float* up_b2   = (const float*)d_in[13];
    const float* tok_w1  = (const float*)d_in[14];
    const float* tok_b1  = (const float*)d_in[15];
    const float* tok_w2  = (const float*)d_in[16];
    const float* tok_b2  = (const float*)d_in[17];
    const int*   eidx    = (const int*)d_in[18];

    const int N = in_sizes[0] / NIN;     // 20000
    const int E = in_sizes[1] / EDIM;    // 320000
    const int* src = eidx;               // edge_index[0]
    const int* dst = eidx + E;           // edge_index[1]

    // ---- workspace partition (all 256B aligned) ----
    char* ws = (char*)d_ws;
    size_t off = 0;
    auto take = [&](size_t bytes) { char* p = ws + off; off = (off + bytes + 255) & ~(size_t)255; return p; };
    bf16_t* hbf   = (bf16_t*)take((size_t)N * HID * 2);
    bf16_t* ebf   = (bf16_t*)take((size_t)E * EDIM * 2);
    float*  agg   = (float*) take((size_t)N * HID * 4);
    bf16_t* nwS   = (bf16_t*)take((size_t)NIN * HID * 2);
    bf16_t* ewS   = (bf16_t*)take((size_t)EDIM * EDIM * 2);
    bf16_t* mw1S  = (bf16_t*)take((size_t)NLAYER * 288 * HID * 2);
    bf16_t* mw2S  = (bf16_t*)take((size_t)NLAYER * HID * HID * 2);
    bf16_t* uw1S  = (bf16_t*)take((size_t)NLAYER * 256 * HID * 2);
    bf16_t* uw2S  = (bf16_t*)take((size_t)NLAYER * HID * HID * 2);

    float* logits = (float*)d_out;
    float* out_h  = logits + N;

    (void)hipFuncSetAttribute((const void*)k_msg,
        hipFuncAttributeMaxDynamicSharedMemorySize, 139264);
    (void)hipFuncSetAttribute((const void*)k_update,
        hipFuncAttributeMaxDynamicSharedMemorySize, 131072);

    auto gsz = [](int total) { return (total + 255) / 256; };

    // ---- weight swizzles (fp32 -> bf16 WMMA B-fragment layout) ----
    k_swizzle<<<gsz(NIN * HID), 256, 0, stream>>>(node_w, nwS, NIN, HID);
    k_swizzle<<<gsz(EDIM * EDIM), 256, 0, stream>>>(edge_w, ewS, EDIM, EDIM);
    for (int l = 0; l < NLAYER; ++l) {
        k_swizzle<<<gsz(288 * HID), 256, 0, stream>>>(msg_w1 + (size_t)l * 288 * HID, mw1S + (size_t)l * 288 * HID, 288, HID);
        k_swizzle<<<gsz(HID * HID), 256, 0, stream>>>(msg_w2 + (size_t)l * HID * HID, mw2S + (size_t)l * HID * HID, HID, HID);
        k_swizzle<<<gsz(256 * HID), 256, 0, stream>>>(up_w1 + (size_t)l * 256 * HID, uw1S + (size_t)l * 256 * HID, 256, HID);
        k_swizzle<<<gsz(HID * HID), 256, 0, stream>>>(up_w2 + (size_t)l * HID * HID, uw2S + (size_t)l * HID * HID, HID, HID);
    }

    // ---- input projections ----
    k_node_proj<<<160, 256, 0, stream>>>(x, nwS, node_b, hbf, N / 16);
    k_edge_proj<<<512, 256, 0, stream>>>(eattr, ewS, edge_b, ebf, E / 16);

    // ---- message-passing layers ----
    for (int l = 0; l < NLAYER; ++l) {
        (void)hipMemsetAsync(agg, 0, (size_t)N * HID * 4, stream);
        k_msg<<<512, 256, 139264, stream>>>(
            hbf, ebf, src, dst,
            mw1S + (size_t)l * 288 * HID, msg_b1 + l * HID,
            mw2S + (size_t)l * HID * HID, msg_b2 + l * HID,
            agg, E / 16);
        k_update<<<160, 256, 131072, stream>>>(
            hbf, agg,
            uw1S + (size_t)l * 256 * HID, up_b1 + l * HID,
            uw2S + (size_t)l * HID * HID, up_b2 + l * HID,
            (l == NLAYER - 1) ? out_h : nullptr, N / 16);
    }

    // ---- token head ----
    k_token<<<gsz(N), 256, 0, stream>>>(out_h, tok_w1, tok_b1, tok_w2, tok_b2, logits, N);
}
